// CausalSelfAttention_6777458393274
// MI455X (gfx1250) — matrix operand order repaired and checked by reading, not compiled, not run on hardware
//
#include <hip/hip_runtime.h>
#include <hip/hip_bf16.h>
#include <math.h>

// ---------------------------------------------------------------------------
// CausalSelfAttention on MI455X (gfx1250).
//   0) pack x, W_qkv, W_out to bf16 once (halves traffic, kills per-tile cvt)
//   1) qkv_bf = x_bf @ W_qkv^T        v_wmma_f32_16x16x32_bf16, bf16 out
//   2) flash attention per (b,h)      WMMA + online softmax, f32 out
//   3) causal depthwise conv K=4 + residual, bf16 out
//   4) out = conv_bf @ W_out^T        WMMA, f32 out
// Tiles staged in LDS with GLOBAL_LOAD_ASYNC_TO_LDS_B128 (ASYNCcnt), double
// buffered. All dims divide tile sizes exactly (B=4 T=2048 C=1024 H=16 D=64).
// ---------------------------------------------------------------------------

typedef __attribute__((ext_vector_type(16))) __bf16 v16bf;
typedef __attribute__((ext_vector_type(8)))  float  v8f;

#define B_    4
#define T_    2048
#define C_    1024
#define H_    16
#define D_    64
#define KCONV 4

// float -> bf16 bits, round-to-nearest-even
static __device__ __forceinline__ unsigned short f2bf(float f) {
  unsigned int u = __float_as_uint(f);
  unsigned int r = u + 0x7FFFu + ((u >> 16) & 1u);
  return (unsigned short)(r >> 16);
}

union FragBF { v16bf v; unsigned int u[8]; unsigned short s[16]; };

// Issue one 16-byte async copy global->LDS (tracked by ASYNCcnt).
static __device__ __forceinline__ void async_b128(unsigned int lds_off,
                                                  const void* gptr) {
  asm volatile("global_load_async_to_lds_b128 %0, %1, off"
               :: "v"(lds_off), "v"((unsigned long long)(uintptr_t)gptr)
               : "memory");
}
static __device__ __forceinline__ void wait_async0() {
  asm volatile("s_wait_asynccnt 0" ::: "memory");
}
static __device__ __forceinline__ void wait_ds0() {
  asm volatile("s_wait_dscnt 0" ::: "memory");
}

// A-fragment (16x32 bf16): lanes 0-15 row=lane, K 0..15;
//                          lanes 16-31 row=lane-16, K 16..31.
static __device__ __forceinline__ v16bf load_frag_a(const unsigned short* base,
                                                    int ld, int mbase, int koff,
                                                    int lane) {
  const int half = lane >> 4;
  const int row  = mbase + (lane & 15);
  const int klo  = half ? 16 : 0;
  const int khi  = klo + 8;
  FragBF f;
  const unsigned int* p = (const unsigned int*)(base + row * ld + koff);
#pragma unroll
  for (int j = 0; j < 4; ++j) f.u[j]     = p[(klo >> 1) + j];
#pragma unroll
  for (int j = 0; j < 4; ++j) f.u[4 + j] = p[(khi >> 1) + j];
  return f.v;
}

// B-fragment (32x16 bf16), source tile "N-major" (row=n, contiguous K):
// lane = N column per half, elems = K 0..15 / 16..31.
static __device__ __forceinline__ v16bf load_frag_b(const unsigned short* base,
                                                    int ld, int nbase, int koff,
                                                    int lane) {
  const int col = nbase + (lane & 15);
  const int kb  = (lane >> 4) ? 16 : 0;
  FragBF f;
  const unsigned int* p = (const unsigned int*)(base + col * ld + koff);
#pragma unroll
  for (int j = 0; j < 8; ++j) f.u[j] = p[(kb >> 1) + j];
  return f.v;
}

// B-fragment from K-major tile (row = k, contiguous N).
static __device__ __forceinline__ v16bf load_frag_b_kmajor(const unsigned short* base,
                                                           int ld, int nbase,
                                                           int lane) {
  const int col = nbase + (lane & 15);
  const int kb  = (lane >> 4) ? 16 : 0;
  FragBF f;
#pragma unroll
  for (int e = 0; e < 16; ++e) f.s[e] = base[(kb + e) * ld + col];
  return f.v;
}

static __device__ __forceinline__ v8f wmma_bf16(v16bf a, v16bf b, v8f c) {
  return __builtin_amdgcn_wmma_f32_16x16x32_bf16(false, a, false, b, (short)0, c,
                                                 false, false);
}

// ---------------------------------------------------------------------------
// f32 -> bf16 pack, 4 elems/thread (b128 load, b64 store)
// ---------------------------------------------------------------------------
__global__ __launch_bounds__(256)
void pack_bf16(const float* __restrict__ s, unsigned short* __restrict__ d) {
  const int i = blockIdx.x * 256 + threadIdx.x;
  const float4 v = ((const float4*)s)[i];
  unsigned long long p = (unsigned long long)f2bf(v.x)
                       | ((unsigned long long)f2bf(v.y) << 16)
                       | ((unsigned long long)f2bf(v.z) << 32)
                       | ((unsigned long long)f2bf(v.w) << 48);
  ((unsigned long long*)d)[i] = p;
}

// ---------------------------------------------------------------------------
// Y[M,N] = A[M,Kd] @ Bm[N,Kd]^T, bf16 operands, f32 accumulate.
// Block tile 64x128, 256 thr = 8 waves (2x4), wave tile 32x32.
// LDS (dynamic): double-buffered  A[64][40] + B[128][40] ushort, 16B-aligned
// rows (80B stride = 20 words -> conflict-free over 16 rows of 64 banks).
// ---------------------------------------------------------------------------
#define GLDA     40          // ushorts per LDS row (A and B)
#define GA_BYTES (64 * 80)   // 5120
#define GB_BYTES (128 * 80)  // 10240
#define GBUF     (GA_BYTES + GB_BYTES)

template <bool OUT_BF16>
__global__ __launch_bounds__(256)
void gemm_bf16(const unsigned short* __restrict__ A,
               const unsigned short* __restrict__ Bm,
               void* __restrict__ Y, int M, int N, int Kd) {
  extern __shared__ char smem[];
  const int tid = threadIdx.x, lane = tid & 31, wave = tid >> 5;
  const int wm = wave >> 2, wn = wave & 3;
  const int row0 = blockIdx.y * 64, col0 = blockIdx.x * 128;

  // issue async copies of one 64x32(A)/128x32(B) bf16 k-slice into buffer `bf`
  auto issue = [&](int k0, int bf) {
    {   // A: 256 chunks of 16B, 1 per thread
      int r = tid >> 2, c = tid & 3;
      async_b128((unsigned)(bf * GBUF + r * 80 + c * 16),
                 A + (size_t)(row0 + r) * Kd + k0 + c * 8);
    }
#pragma unroll
    for (int i = 0; i < 2; ++i) {  // B: 512 chunks, 2 per thread
      int id = tid + i * 256; int r = id >> 2, c = id & 3;
      async_b128((unsigned)(bf * GBUF + GA_BYTES + r * 80 + c * 16),
                 Bm + (size_t)(col0 + r) * Kd + k0 + c * 8);
    }
  };

  const v8f vzero = {0.f, 0.f, 0.f, 0.f, 0.f, 0.f, 0.f, 0.f};
  v8f acc[2][2];
#pragma unroll
  for (int i = 0; i < 2; ++i)
#pragma unroll
    for (int j = 0; j < 2; ++j) acc[i][j] = vzero;

  issue(0, 0);
  wait_async0();
  __syncthreads();

  for (int k0 = 0; k0 < Kd; k0 += 32) {
    const int cur = (k0 >> 5) & 1;
    if (k0 + 32 < Kd) issue(k0 + 32, cur ^ 1);   // prefetch next slice

    const unsigned short* As = (const unsigned short*)(smem + cur * GBUF);
    const unsigned short* Bs = (const unsigned short*)(smem + cur * GBUF + GA_BYTES);
    v16bf a0 = load_frag_a(As, GLDA, wm * 32,      0, lane);
    v16bf a1 = load_frag_a(As, GLDA, wm * 32 + 16, 0, lane);
    v16bf b0 = load_frag_b(Bs, GLDA, wn * 32,      0, lane);
    v16bf b1 = load_frag_b(Bs, GLDA, wn * 32 + 16, 0, lane);
    acc[0][0] = wmma_bf16(a0, b0, acc[0][0]);
    acc[0][1] = wmma_bf16(a0, b1, acc[0][1]);
    acc[1][0] = wmma_bf16(a1, b0, acc[1][0]);
    acc[1][1] = wmma_bf16(a1, b1, acc[1][1]);

    wait_async0();      // next slice resident
    __syncthreads();    // all waves done reading cur before it is reused
  }

  const int half = lane >> 4, nlane = lane & 15;
#pragma unroll
  for (int i = 0; i < 2; ++i)
#pragma unroll
    for (int j = 0; j < 2; ++j)
#pragma unroll
      for (int r = 0; r < 8; ++r) {
        int m = row0 + wm * 32 + 16 * i + r + half * 8;
        int n = col0 + wn * 32 + 16 * j + nlane;
        if (OUT_BF16) ((unsigned short*)Y)[(size_t)m * N + n] = f2bf(acc[i][j][r]);
        else          ((float*)Y)[(size_t)m * N + n]          = acc[i][j][r];
      }
}

// ---------------------------------------------------------------------------
// Flash attention: block = (b,h) x 64-query tile; 4 waves x 16 rows; 128 thr.
// qkv_bf layout [B,T,3C] bf16 (Q +0, K +C, V +2C). K/V double buffered via
// async-to-LDS. Output f32 (feeds conv residual).
// LDS (dynamic): Qs 64x72u (144B rows), K/V 2x(32x72u), Ps 4x16x40u.
// ---------------------------------------------------------------------------
#define FLDQ    72
#define FQ_OFF  0
#define FQ_B    (64 * 144)              // 9216
#define FKV_B   (32 * 144)              // 4608
#define FK_OFF(b) (FQ_B + (b) * 2 * FKV_B)
#define FV_OFF(b) (FK_OFF(b) + FKV_B)
#define FP_OFF  (FQ_B + 4 * FKV_B)      // 27648
#define FLDP    40

__global__ __launch_bounds__(128)
void flash_attn(const unsigned short* __restrict__ qkv, float* __restrict__ out) {
  extern __shared__ char smem[];
  const int tid = threadIdx.x, lane = tid & 31, wave = tid >> 5;
  const int b = blockIdx.y >> 4, h = blockIdx.y & 15;
  const int qbase = blockIdx.x * 64;
  const int hoff = h * D_;
  const size_t ldq = 3 * C_;

  auto issue_kv = [&](int kv, int bf) {
#pragma unroll
    for (int i = 0; i < 2; ++i) {   // K and V: 256 chunks each total
      int id = tid + i * 128; int r = id >> 3, c = id & 7;
      size_t ro = (size_t)(b * T_ + kv + r) * ldq + hoff + c * 8;
      async_b128((unsigned)(FK_OFF(bf) + r * 144 + c * 16), qkv + ro + C_);
      async_b128((unsigned)(FV_OFF(bf) + r * 144 + c * 16), qkv + ro + 2 * C_);
    }
  };

  // Q tile 64x64: 512 chunks, 4 per thread
#pragma unroll
  for (int i = 0; i < 4; ++i) {
    int id = tid + i * 128; int r = id >> 3, c = id & 7;
    async_b128((unsigned)(FQ_OFF + r * 144 + c * 16),
               qkv + (size_t)(b * T_ + qbase + r) * ldq + hoff + c * 8);
  }
  issue_kv(0, 0);
  wait_async0();
  __syncthreads();

  const unsigned short* Qs = (const unsigned short*)(smem + FQ_OFF);
  unsigned short*       Ps = (unsigned short*)(smem + FP_OFF);
  const int mbase = wave * 16;
  const int half = lane >> 4, nlane = lane & 15;
  const v8f vzero = {0.f, 0.f, 0.f, 0.f, 0.f, 0.f, 0.f, 0.f};
  float m_i[8], l_i[8];
  v8f o[4];
#pragma unroll
  for (int r = 0; r < 8; ++r) { m_i[r] = -__builtin_inff(); l_i[r] = 0.f; }
#pragma unroll
  for (int n = 0; n < 4; ++n) o[n] = vzero;

  const int jmax = (qbase + 63) >> 5;           // causal kv-tile bound
  for (int j = 0; j <= jmax; ++j) {
    const int cur = j & 1;
    const int kv = j * 32;
    if (j < jmax) issue_kv(kv + 32, cur ^ 1);   // prefetch next K/V tile

    const unsigned short* Ks = (const unsigned short*)(smem + FK_OFF(cur));
    const unsigned short* Vs = (const unsigned short*)(smem + FV_OFF(cur));

    // S(16x32) = Q(16x64) @ K^T
    v8f s[2]; s[0] = vzero; s[1] = vzero;
#pragma unroll
    for (int kd = 0; kd < 64; kd += 32) {
      v16bf aq  = load_frag_a(Qs, FLDQ, mbase, kd, lane);
      v16bf bk0 = load_frag_b(Ks, FLDQ, 0,  kd, lane);
      v16bf bk1 = load_frag_b(Ks, FLDQ, 16, kd, lane);
      s[0] = wmma_bf16(aq, bk0, s[0]);
      s[1] = wmma_bf16(aq, bk1, s[1]);
    }

    // scale + causal mask + online softmax (row r lives across one lane half)
#pragma unroll
    for (int r = 0; r < 8; ++r) {
      const int qg = qbase + mbase + r + half * 8;
      float s0 = s[0][r] * 0.125f;  if (kv + nlane      > qg) s0 = -__builtin_inff();
      float s1 = s[1][r] * 0.125f;  if (kv + 16 + nlane > qg) s1 = -__builtin_inff();
      float mx = fmaxf(s0, s1);
      mx = fmaxf(mx, __shfl_xor(mx, 1, 32));
      mx = fmaxf(mx, __shfl_xor(mx, 2, 32));
      mx = fmaxf(mx, __shfl_xor(mx, 4, 32));
      mx = fmaxf(mx, __shfl_xor(mx, 8, 32));
      const float mnew  = fmaxf(m_i[r], mx);
      const float alpha = __expf(m_i[r] - mnew);
      const float p0 = __expf(s0 - mnew), p1 = __expf(s1 - mnew);
      float sum = p0 + p1;
      sum += __shfl_xor(sum, 1, 32);
      sum += __shfl_xor(sum, 2, 32);
      sum += __shfl_xor(sum, 4, 32);
      sum += __shfl_xor(sum, 8, 32);
      l_i[r] = l_i[r] * alpha + sum;
      m_i[r] = mnew;
#pragma unroll
      for (int n = 0; n < 4; ++n) o[n][r] *= alpha;
      Ps[(wave * 16 + r + half * 8) * FLDP + nlane]      = f2bf(p0);
      Ps[(wave * 16 + r + half * 8) * FLDP + 16 + nlane] = f2bf(p1);
    }
    wait_ds0();   // intra-wave: P stores complete before re-layout loads

    // O += P(16x32) @ V(32x64)
    v16bf ap = load_frag_a(Ps + wave * 16 * FLDP, FLDP, 0, 0, lane);
#pragma unroll
    for (int n = 0; n < 4; ++n) {
      v16bf bv = load_frag_b_kmajor(Vs, FLDQ, n * 16, lane);
      o[n] = wmma_bf16(ap, bv, o[n]);
    }

    wait_async0();     // next K/V tile resident
    __syncthreads();   // everyone done with cur before reuse
  }

#pragma unroll
  for (int n = 0; n < 4; ++n)
#pragma unroll
    for (int r = 0; r < 8; ++r) {
      const int qg = qbase + mbase + r + half * 8;
      out[(size_t)(b * T_ + qg) * C_ + hoff + n * 16 + nlane] = o[n][r] / l_i[r];
    }
}

// ---------------------------------------------------------------------------
// y_bf16 = bf16( x + causal_depthwise_conv_K4(x) + bias ), x is f32
// ---------------------------------------------------------------------------
__global__ __launch_bounds__(256)
void conv_residual(const float* __restrict__ x, const float* __restrict__ w,
                   const float* __restrict__ bias, unsigned short* __restrict__ y) {
  const size_t idx = (size_t)blockIdx.x * 256 + threadIdx.x;   // [B,T,C] flat
  const int c = (int)(idx & (C_ - 1));
  const int t = (int)((idx >> 10) & (T_ - 1));
  const float* xp = x + idx;
  float acc = bias[c];
#pragma unroll
  for (int k = 0; k < KCONV; ++k) {
    const int ts = t - (KCONV - 1) + k;
    if (ts >= 0) acc += xp[(ts - t) * C_] * w[c * KCONV + k];
  }
  y[idx] = f2bf(xp[0] + acc);
}

// ---------------------------------------------------------------------------
extern "C" void kernel_launch(void* const* d_in, const int* in_sizes, int n_in,
                              void* d_out, int out_size, void* d_ws, size_t ws_size,
                              hipStream_t stream) {
  const float* x      = (const float*)d_in[0];
  const float* W_qkv  = (const float*)d_in[1];
  const float* W_out  = (const float*)d_in[2];
  const float* conv_w = (const float*)d_in[3];
  const float* conv_b = (const float*)d_in[4];
  float* out = (float*)d_out;

  const size_t MT = (size_t)B_ * T_;       // 8192
  // workspace layout (all 16B aligned): ~126 MB total
  unsigned short* qkv_bf  = (unsigned short*)d_ws;            // MT*3C
  float*          attn    = (float*)(qkv_bf + MT * 3 * C_);   // MT*C f32
  unsigned short* conv_bf = (unsigned short*)(attn + MT * C_);// MT*C
  unsigned short* x_bf    = conv_bf + MT * C_;                // MT*C
  unsigned short* wqkv_bf = x_bf + MT * C_;                   // 3C*C
  unsigned short* wout_bf = wqkv_bf + (size_t)3 * C_ * C_;    // C*C

  // 0) one-time bf16 packs
  pack_bf16<<<(MT * C_ / 4) / 256, 256, 0, stream>>>(x, x_bf);
  pack_bf16<<<((size_t)3 * C_ * C_ / 4) / 256, 256, 0, stream>>>(W_qkv, wqkv_bf);
  pack_bf16<<<((size_t)C_ * C_ / 4) / 256, 256, 0, stream>>>(W_out, wout_bf);

  // 1) QKV projection -> bf16
  {
    dim3 g(3 * C_ / 128, MT / 64);
    gemm_bf16<true><<<g, 256, 2 * GBUF, stream>>>(x_bf, wqkv_bf, qkv_bf,
                                                  (int)MT, 3 * C_, C_);
  }
  // 2) causal flash attention -> f32
  {
    dim3 g(T_ / 64, B_ * H_);
    flash_attn<<<g, 128, FP_OFF + 4 * 16 * FLDP * 2, stream>>>(qkv_bf, attn);
  }
  // 3) depthwise conv + residual -> bf16
  conv_residual<<<(B_ * T_ * C_) / 256, 256, 0, stream>>>(attn, conv_w, conv_b,
                                                          conv_bf);
  // 4) output projection -> f32 d_out
  {
    dim3 g(C_ / 128, MT / 64);
    gemm_bf16<false><<<g, 256, 2 * GBUF, stream>>>(conv_bf, wout_bf, out,
                                                   (int)MT, C_, C_);
  }
}
